// OptRSSM_36575941492751
// MI455X (gfx1250) — matrix operand order repaired and checked by reading, not compiled, not run on hardware
//
#include <hip/hip_runtime.h>

#define T_   64
#define B_   512
#define OBS_ 1024
#define ACT_ 32
#define H_   1024
#define D_   1024
#define S_   1024
#define NT_  32

typedef __attribute__((ext_vector_type(16))) __bf16 v16bf;
typedef __attribute__((ext_vector_type(8)))  float  v8f;

union FragCvt { uint4 q[2]; v16bf v; };

static __device__ __forceinline__ __bf16 f2bf(float f) {
  unsigned u = __float_as_uint(f);
  u += 0x7FFFu + ((u >> 16) & 1u);              // round-to-nearest-even
  unsigned short s = (unsigned short)(u >> 16);
  return __builtin_bit_cast(__bf16, s);
}

// raw 32-bit LDS byte offset from a generic pointer to a __shared__ object
typedef __attribute__((address_space(3))) const void lds_cv_t;
static __device__ __forceinline__ unsigned lds_off(const void* p) {
  return (unsigned)(unsigned long long)(lds_cv_t*)p;
}

// async copy of 32 contiguous bytes (2 x b128) global -> LDS, per lane
static __device__ __forceinline__ void async_cp32(unsigned lds, const void* gp) {
  asm volatile("global_load_async_to_lds_b128 %0, %1, off"
               :: "v"(lds), "v"(gp) : "memory");
  asm volatile("global_load_async_to_lds_b128 %0, %1, off offset:16"
               :: "v"(lds), "v"(gp) : "memory");
}

// ---------------------------------------------------------------- converters
__global__ __launch_bounds__(256) void cvt_k(const float* __restrict__ in,
                                             __bf16* __restrict__ out, int n) {
  int i = (blockIdx.x * 256 + threadIdx.x) * 4;
  if (i < n) {
#pragma unroll
    for (int j = 0; j < 4; ++j) out[i + j] = f2bf(in[i + j]);
  }
}

__global__ __launch_bounds__(256) void init_state(const float* __restrict__ h0d,
                                                  const float* __restrict__ h0s,
                                                  float* __restrict__ deter_state,
                                                  __bf16* __restrict__ xcat,
                                                  __bf16* __restrict__ stoch_bf) {
  int i = blockIdx.x * 256 + threadIdx.x;    // over B*D == B*S
  if (i < B_ * D_) {
    float d = h0d[i];
    deter_state[i] = d;
    int b = i >> 10, c = i & 1023;
    xcat[(size_t)b * (H_ + D_) + H_ + c] = f2bf(d);
    stoch_bf[i] = f2bf(h0s[i]);
  }
}

// ---------------------------------------------------------------- WMMA GEMM
// Y[M,N] = epi( sum_k A0[m,k]*W0[n,k] + sum_k A1[m,k]*W1[n,k] + bias[n] )
// A row-major [M,K], W row-major [N,K] (== column-major B operand).
// Block tile 128x128, 8 waves (2 M x 4 N), each wave 64x32 = 4x2 WMMA tiles.
// 3-stage ASYNCcnt pipeline with compile-time stage indices (modulo-3 unroll),
// one barrier per K-step.
template <bool ELU>
__global__ __launch_bounds__(256) void gemm_wmma(
    const __bf16* __restrict__ A0, const __bf16* __restrict__ W0, int K0,
    const __bf16* __restrict__ A1, const __bf16* __restrict__ W1, int K1,
    const float* __restrict__ bias,
    float* __restrict__ outF, __bf16* __restrict__ outB, int ldB, int N) {
  __shared__ __align__(16) __bf16 lsA[3][128 * 40];   // 32 cols + 8 pad, x3 stages
  __shared__ __align__(16) __bf16 lsB[3][128 * 40];

  const int tid  = threadIdx.x;
  const int lane = tid & 31;
  const int wave = tid >> 5;
  const int wn   = wave & 3;    // N strip (32 wide)
  const int wm   = wave >> 2;   // M strip (64 tall)
  const int bm   = blockIdx.y * 128;
  const int bn   = blockIdx.x * 128;

  v8f acc[4][2] = {};

  const int ldrow = tid >> 1;          // 0..127
  const int ldcol = (tid & 1) * 16;    // 0 / 16
  const int nt0 = K0 >> 5;
  const int nt1 = K1 >> 5;
  const int ntot = nt0 + nt1;

  // hoisted per-thread bases (issue = select + offset add)
  const __bf16* baseA0 = A0 + (size_t)(bm + ldrow) * K0 + ldcol;
  const __bf16* baseW0 = W0 + (size_t)(bn + ldrow) * K0 + ldcol;
  const __bf16* baseA1 = A1 ? A1 + (size_t)(bm + ldrow) * K1 + ldcol : nullptr;
  const __bf16* baseW1 = W1 ? W1 + (size_t)(bn + ldrow) * K1 + ldcol : nullptr;
  const unsigned laddrA = lds_off(&lsA[0][ldrow * 40 + ldcol]);
  const unsigned laddrB = lds_off(&lsB[0][ldrow * 40 + ldcol]);
  const unsigned stageBytes = 128 * 40 * 2;

  auto issue_tile = [&](int ti, int st) {   // st is always a literal at call sites
    const __bf16* ga;
    const __bf16* gw;
    if (ti < nt0) { ga = baseA0 + ti * 32;          gw = baseW0 + ti * 32; }
    else          { ga = baseA1 + (ti - nt0) * 32;  gw = baseW1 + (ti - nt0) * 32; }
    async_cp32(laddrA + (unsigned)st * stageBytes, ga);
    async_cp32(laddrB + (unsigned)st * stageBytes, gw);
  };

  const int nn = lane & 15;
  const int kb = (lane >> 4) * 16;
  const int rr = lane & 15;
  const int kq = (lane >> 4) * 8;

  auto compute = [&](int st) {              // st literal -> const LDS offsets
    // B fragments: lane holds col n=lane&15, K base (lane>>4)*16, 16 contiguous
    v16bf bf[2];
#pragma unroll
    for (int nt = 0; nt < 2; ++nt) {
      const int row = wn * 32 + nt * 16 + nn;
      FragCvt f;
      f.q[0] = *(const uint4*)&lsB[st][row * 40 + kb];
      f.q[1] = *(const uint4*)&lsB[st][row * 40 + kb + 8];
      bf[nt] = f.v;
    }
    // A fragments: lane holds row m=lane&15, K base (lane>>4)*16, 16 contiguous
#pragma unroll
    for (int mt = 0; mt < 4; ++mt) {
      const int row = wm * 64 + mt * 16 + rr;
      FragCvt f;
      f.q[0] = *(const uint4*)&lsA[st][row * 40 + kb];
      f.q[1] = *(const uint4*)&lsA[st][row * 40 + kb + 8];
      v16bf af = f.v;
#pragma unroll
      for (int nt = 0; nt < 2; ++nt)
        acc[mt][nt] = __builtin_amdgcn_wmma_f32_16x16x32_bf16(
            false, af, false, bf[nt], (short)0, acc[mt][nt], false, false);
    }
  };

  auto step_steady = [&](int ti, int cs, int ps) {
    asm volatile("s_wait_asynccnt 0x4" ::: "memory");
    __syncthreads();                 // block-wide: tile ti landed everywhere
    issue_tile(ti + 2, ps);          // overwrites stage read in previous iter (safe after barrier)
    compute(cs);
  };
  auto step_tail = [&](bool last, int cs) {
    if (!last) asm volatile("s_wait_asynccnt 0x4" ::: "memory");
    else       asm volatile("s_wait_asynccnt 0x0" ::: "memory");
    __syncthreads();
    compute(cs);
  };

  issue_tile(0, 0);
  if (ntot > 1) issue_tile(1, 1);

  const int steady = ntot - 2;       // >= 0 for all shapes used here
  int ti = 0;
  // groups of 3 -> stage pairs (0,2), (1,0), (2,1) are compile-time
  for (; ti + 3 <= steady; ti += 3) {
    step_steady(ti,     0, 2);
    step_steady(ti + 1, 1, 0);
    step_steady(ti + 2, 2, 1);
  }
  const int rem = steady - ti;       // 0..2, and ti % 3 == 0 here
  if (rem > 0) step_steady(ti,     0, 2);
  if (rem > 1) step_steady(ti + 1, 1, 0);
  // tail: 2 iterations; compute stages = steady%3, (steady+1)%3 (literals per case)
  if (ntot >= 2) {
    switch (rem) {                   // rem == steady % 3
      case 0:  step_tail(false, 0); step_tail(true, 1); break;
      case 1:  step_tail(false, 1); step_tail(true, 2); break;
      default: step_tail(false, 2); step_tail(true, 0); break;
    }
  } else if (ntot == 1) {
    step_tail(true, 0);
  }

  // epilogue: C/D layout -> lane&15 = n, VGPR j: m = (lane>>4)*8 + j
  const int nloc  = lane & 15;
  const int mhalf = (lane >> 4) * 8;
#pragma unroll
  for (int mt = 0; mt < 4; ++mt) {
#pragma unroll
    for (int nt = 0; nt < 2; ++nt) {
      const int n = bn + wn * 32 + nt * 16 + nloc;
      const float bv = bias ? bias[n] : 0.f;
#pragma unroll
      for (int j = 0; j < 8; ++j) {
        const int m = bm + wm * 64 + mt * 16 + mhalf + j;
        float v = acc[mt][nt][j] + bv;
        if (ELU) v = v > 0.f ? v : (__expf(v) - 1.f);
        if (outF) outF[(size_t)m * N + n] = v;
        if (outB) outB[(size_t)m * ldB + n] = f2bf(v);
      }
    }
  }
}

// ------------------------------------------------- LayerNorm + GRU gate fuse
__global__ __launch_bounds__(256) void gru_ln(
    const float* __restrict__ parts,              // [B, 3072]
    const float* __restrict__ lns, const float* __restrict__ lnb,
    float* __restrict__ deter_state,              // [B, D] prev -> new
    float* __restrict__ out_deter,                // d_out deters[t]
    __bf16* __restrict__ out_deter_bf,            // deters_bf[t]
    __bf16* __restrict__ xcat) {                  // cols 1024..2047
  const int b = blockIdx.x, tid = threadIdx.x;
  const float* row = parts + (size_t)b * 3072;
  float v[12];
  float s = 0.f, ss = 0.f;
#pragma unroll
  for (int i = 0; i < 12; ++i) {
    float x = row[tid + 256 * i];
    v[i] = x; s += x; ss += x * x;
  }
#pragma unroll
  for (int o = 16; o; o >>= 1) { s += __shfl_xor(s, o, 32); ss += __shfl_xor(ss, o, 32); }
  __shared__ float shs[8], shss[8];
  if ((tid & 31) == 0) { shs[tid >> 5] = s; shss[tid >> 5] = ss; }
  __syncthreads();
  float ts = 0.f, tss = 0.f;
#pragma unroll
  for (int w = 0; w < 8; ++w) { ts += shs[w]; tss += shss[w]; }
  const float mean = ts * (1.f / 3072.f);
  const float var  = tss * (1.f / 3072.f) - mean * mean;
  const float rstd = rsqrtf(var + 1e-5f);
#pragma unroll
  for (int j = 0; j < 4; ++j) {
    const int d = tid + 256 * j;
    float g0 = (v[j]     - mean) * rstd * lns[d]        + lnb[d];
    float g1 = (v[4 + j] - mean) * rstd * lns[1024 + d] + lnb[1024 + d];
    float g2 = (v[8 + j] - mean) * rstd * lns[2048 + d] + lnb[2048 + d];
    const float reset  = 1.f / (1.f + __expf(-g0));
    const float cand   = tanhf(reset * g1);
    const float update = 1.f / (1.f + __expf(-(g2 - 1.f)));   // update_bias = -1
    const size_t idx = (size_t)b * D_ + d;
    const float dprev = deter_state[idx];
    const float dnew  = update * cand + (1.f - update) * dprev;
    deter_state[idx]  = dnew;
    out_deter[idx]    = dnew;
    out_deter_bf[idx] = f2bf(dnew);
    xcat[(size_t)b * (H_ + D_) + H_ + d] = f2bf(dnew);
  }
}

// ------------------------------------------- Gumbel-max sampler (1 wave / dist)
__global__ __launch_bounds__(256) void sample_k(
    const float* __restrict__ post, const float* __restrict__ unif,
    float* __restrict__ stoch_out, __bf16* __restrict__ stoch_bf) {
  const int gid  = blockIdx.x * 256 + threadIdx.x;
  const int lane = gid & 31;
  const int row  = gid >> 5;                     // b*NT + nt
  const size_t off = (size_t)row * 32 + lane;
  const float x = post[off];
  float m = x;
#pragma unroll
  for (int o = 16; o; o >>= 1) m = fmaxf(m, __shfl_xor(m, o, 32));
  float e = __expf(x - m);
  float se = e;
#pragma unroll
  for (int o = 16; o; o >>= 1) se += __shfl_xor(se, o, 32);
  const float prob = e / se;
  const float EPS = 1.1920929e-07f;
  float u = fminf(fmaxf(unif[off], EPS), 1.f - EPS);
  const float g = x - __logf(-__logf(u));
  float gm = g;
#pragma unroll
  for (int o = 16; o; o >>= 1) gm = fmaxf(gm, __shfl_xor(gm, o, 32));
  int cand = (g == gm) ? lane : 32;              // first max wins (argmax tie rule)
#pragma unroll
  for (int o = 16; o; o >>= 1) cand = min(cand, __shfl_xor(cand, o, 32));
  const float onehot = (lane == cand) ? 1.f : 0.f;
  const float sv = onehot + prob - prob;         // straight-through (fwd value)
  stoch_out[off] = sv;
  stoch_bf[off]  = f2bf(sv);
}

// ---------------------------------------------------------------------- host
extern "C" void kernel_launch(void* const* d_in, const int* in_sizes, int n_in,
                              void* d_out, int out_size, void* d_ws, size_t ws_size,
                              hipStream_t stream) {
  (void)in_sizes; (void)n_in; (void)out_size; (void)ws_size;
  const float* obs         = (const float*)d_in[0];
  const float* act         = (const float*)d_in[1];
  const float* h0_deter    = (const float*)d_in[2];
  const float* h0_stoch    = (const float*)d_in[3];
  const float* unif        = (const float*)d_in[4];
  const float* W_obs_o     = (const float*)d_in[5];
  const float* W_img_a     = (const float*)d_in[6];
  const float* W_img_s     = (const float*)d_in[7];
  const float* b_img_s     = (const float*)d_in[8];
  const float* W_gru       = (const float*)d_in[9];
  const float* ln_scale    = (const float*)d_in[10];
  const float* ln_bias     = (const float*)d_in[11];
  const float* W_img_out   = (const float*)d_in[12];
  const float* b_img_out   = (const float*)d_in[13];
  const float* W_img_proj  = (const float*)d_in[14];
  const float* b_img_proj  = (const float*)d_in[15];
  const float* W_obs_out_d = (const float*)d_in[16];
  const float* b_obs_out_d = (const float*)d_in[17];
  const float* W_obs_proj  = (const float*)d_in[18];
  const float* b_obs_proj  = (const float*)d_in[19];

  float* d_deters = (float*)d_out;
  float* d_stochs = d_deters + (size_t)T_ * B_ * D_;
  float* d_posts  = d_stochs + (size_t)T_ * B_ * S_;
  float* d_priors = d_posts  + (size_t)T_ * B_ * S_;

  // workspace carve (~242 MB total)
  char* wp = (char*)d_ws;
  auto abf  = [&](size_t n) { __bf16* r = (__bf16*)wp; wp += (n * 2 + 255) & ~(size_t)255; return r; };
  auto af32 = [&](size_t n) { float*  r = (float*) wp; wp += (n * 4 + 255) & ~(size_t)255; return r; };

  __bf16* Wobso_bf    = abf((size_t)H_ * OBS_);
  __bf16* Wimga_bf    = abf((size_t)H_ * ACT_);
  __bf16* Wimgs_bf    = abf((size_t)H_ * S_);
  __bf16* Wgru_bf     = abf((size_t)3 * D_ * (H_ + D_));
  __bf16* Wimgout_bf  = abf((size_t)H_ * D_);
  __bf16* Wimgproj_bf = abf((size_t)S_ * H_);
  __bf16* Wobsoutd_bf = abf((size_t)H_ * D_);
  __bf16* Wobsproj_bf = abf((size_t)S_ * H_);
  __bf16* obs_bf      = abf((size_t)T_ * B_ * OBS_);
  __bf16* act_bf      = abf((size_t)T_ * B_ * ACT_);
  __bf16* xcat        = abf((size_t)B_ * (H_ + D_));
  __bf16* stoch_bf    = abf((size_t)B_ * S_);
  __bf16* h_bf        = abf((size_t)B_ * H_);
  __bf16* deters_bf   = abf((size_t)T_ * B_ * D_);
  __bf16* hp_bf       = abf((size_t)T_ * B_ * H_);
  float*  parts       = af32((size_t)B_ * 3 * D_);
  float*  deter_st    = af32((size_t)B_ * D_);

  auto cvt = [&](const float* in, __bf16* outp, size_t n) {
    cvt_k<<<(int)((n / 4 + 255) / 256), 256, 0, stream>>>(in, outp, (int)n);
  };
  cvt(W_obs_o, Wobso_bf, (size_t)H_ * OBS_);
  cvt(W_img_a, Wimga_bf, (size_t)H_ * ACT_);
  cvt(W_img_s, Wimgs_bf, (size_t)H_ * S_);
  cvt(W_gru, Wgru_bf, (size_t)3 * D_ * (H_ + D_));
  cvt(W_img_out, Wimgout_bf, (size_t)H_ * D_);
  cvt(W_img_proj, Wimgproj_bf, (size_t)S_ * H_);
  cvt(W_obs_out_d, Wobsoutd_bf, (size_t)H_ * D_);
  cvt(W_obs_proj, Wobsproj_bf, (size_t)S_ * H_);
  cvt(obs, obs_bf, (size_t)T_ * B_ * OBS_);
  cvt(act, act_bf, (size_t)T_ * B_ * ACT_);

  init_state<<<(B_ * D_ + 255) / 256, 256, 0, stream>>>(h0_deter, h0_stoch,
                                                        deter_st, xcat, stoch_bf);

  auto gemm = [&](bool elu, const __bf16* A0c, const __bf16* W0c, int K0,
                  const __bf16* A1c, const __bf16* W1c, int K1,
                  const float* biasc, float* oF, __bf16* oB, int ldB, int M, int N) {
    dim3 g(N / 128, M / 128), blk(256);
    if (elu) gemm_wmma<true><<<g, blk, 0, stream>>>(A0c, W0c, K0, A1c, W1c, K1, biasc, oF, oB, ldB, N);
    else     gemm_wmma<false><<<g, blk, 0, stream>>>(A0c, W0c, K0, A1c, W1c, K1, biasc, oF, oB, ldB, N);
  };

  for (int t = 0; t < T_; ++t) {
    // x = elu(stoch@W_img_s^T + act[t]@W_img_a^T + b)  -> xcat[:, 0:1024] bf16
    gemm(true, stoch_bf, Wimgs_bf, S_, act_bf + (size_t)t * B_ * ACT_, Wimga_bf, ACT_,
         b_img_s, nullptr, xcat, H_ + D_, B_, H_);
    // parts = [x|deter] @ W_gru^T  (raw, LN applied next)
    gemm(false, xcat, Wgru_bf, H_ + D_, nullptr, nullptr, 0,
         nullptr, parts, nullptr, 0, B_, 3 * D_);
    // LN + gates + deter update
    gru_ln<<<B_, 256, 0, stream>>>(parts, ln_scale, ln_bias, deter_st,
                                   d_deters + (size_t)t * B_ * D_,
                                   deters_bf + (size_t)t * B_ * D_, xcat);
    // h = elu(deter@W_obs_out_d^T + obs[t]@W_obs_o^T + b)
    gemm(true, deters_bf + (size_t)t * B_ * D_, Wobsoutd_bf, D_,
         obs_bf + (size_t)t * B_ * OBS_, Wobso_bf, OBS_,
         b_obs_out_d, nullptr, h_bf, H_, B_, H_);
    // post = h@W_obs_proj^T + b
    gemm(false, h_bf, Wobsproj_bf, H_, nullptr, nullptr, 0,
         b_obs_proj, d_posts + (size_t)t * B_ * S_, nullptr, 0, B_, S_);
    // Gumbel-max sample -> stochs[t], stoch_bf
    sample_k<<<(B_ * NT_ * 32) / 256, 256, 0, stream>>>(
        d_posts + (size_t)t * B_ * S_, unif + (size_t)t * B_ * S_,
        d_stochs + (size_t)t * B_ * S_, stoch_bf);
  }

  // priors path (time-parallel)
  gemm(true, deters_bf, Wimgout_bf, D_, nullptr, nullptr, 0,
       b_img_out, nullptr, hp_bf, H_, T_ * B_, H_);
  gemm(false, hp_bf, Wimgproj_bf, H_, nullptr, nullptr, 0,
       b_img_proj, d_priors, nullptr, 0, T_ * B_, S_);
}
